// MixtureOfExpertsModel_72919954752179
// MI455X (gfx1250) — matrix operand order repaired and checked by reading, not compiled, not run on hardware
//
#include <hip/hip_runtime.h>

typedef __attribute__((ext_vector_type(16))) _Float16 v16h;
typedef __attribute__((ext_vector_type(8)))  float    v8f;
typedef unsigned int u32;

union UF { v16h v; u32 u[8]; _Float16 h[16]; };

static __device__ inline v8f vzero8() {
  v8f z;
#pragma unroll
  for (int i = 0; i < 8; ++i) z[i] = 0.f;
  return z;
}

// CDNA5 async global->LDS copy (16B per lane), GVS addressing:
//   mem = SADDR(64b sgpr pair) + VADDR(32b vgpr byte offset)
//   lds = LDS_BASE + VDST(32b vgpr byte offset)
static __device__ inline void async_copy_b128(u32 lds_byte_off, u32 g_byte_off,
                                              const void* gbase) {
  asm volatile("global_load_async_to_lds_b128 %0, %1, %2"
               :
               : "v"(lds_byte_off), "v"(g_byte_off),
                 "s"((unsigned long long)(uintptr_t)gbase)
               : "memory");
}
static __device__ inline void wait_async0() {
  asm volatile("s_wait_asynccnt 0" ::: "memory");
}
// Truncate a generic pointer to the LDS byte offset (LDS aperture lives in
// the high dword; low dword is the in-LDS offset).
static __device__ inline u32 lds_off(const void* p) {
  return (u32)(uintptr_t)p;
}

#define BN_EPS 1e-3f

// ---------------------------------------------------------------------------
// Kernel 1: conv1 (3x3, 3->32, SAME) + bias + relu + BN + 2x2 maxpool, fused.
// One block per image. Implicit GEMM: M = pixels (16/tile), K = 27 (pad 32),
// N = 32 channels (2 n-tiles). Output h1: [2048,16,16,32] f16.
// ---------------------------------------------------------------------------
__global__ __launch_bounds__(256) void conv1_kernel(
    const float* __restrict__ x, const float* __restrict__ w,
    const float* __restrict__ bias, const float* __restrict__ g,
    const float* __restrict__ bb, const float* __restrict__ bm,
    const float* __restrict__ bv, _Float16* __restrict__ h1) {
  __shared__ _Float16 lX[34 * 34 * 3];   // zero-padded input patch (f16)
  __shared__ _Float16 lWt[32 * 32];      // weights transposed [o][k], k padded

  const int tid = threadIdx.x;
  const int img = blockIdx.x;

  for (int i = tid; i < 34 * 34 * 3; i += 256) lX[i] = (_Float16)0.f;
  for (int i = tid; i < 32 * 32; i += 256) lWt[i] = (_Float16)0.f;
  __syncthreads();

  const float* xi = x + (size_t)img * 32 * 32 * 3;
  for (int i = tid; i < 32 * 32 * 3; i += 256) {
    int c = i % 3, p = i / 3;
    int xx = p & 31, yy = p >> 5;
    lX[((yy + 1) * 34 + (xx + 1)) * 3 + c] = (_Float16)xi[i];
  }
  // cW1 is HWIO [3,3,3,32] -> flat index = k*32 + o with k = (dy*3+dx)*3+c
  for (int i = tid; i < 27 * 32; i += 256) {
    int k = i >> 5, o = i & 31;
    lWt[o * 32 + k] = (_Float16)w[i];
  }
  __syncthreads();

  const int wv = tid >> 5, lane = tid & 31;
  const int lr = lane & 15, half = lane >> 4;
  const int kbA = half ? 8 : 0;    // A fragment K base (16-bit A layout)
  const int kbB = half ? 16 : 0;   // unused; B fragments use kbA, same k order as A

  UF bf[2];
#pragma unroll
  for (int nt = 0; nt < 2; ++nt) {
    int o = nt * 16 + lr;
#pragma unroll
    for (int t = 0; t < 8; ++t)
      bf[nt].u[t] = *(const u32*)&lWt[o * 32 + ((t < 4) ? (kbA + 2 * t) : (16 + kbA + 2 * (t - 4)))];
  }

#pragma unroll
  for (int p = 0; p < 2; ++p) {         // y-pair within this wave's 4 rows
    const int y0 = wv * 4 + 2 * p;
#pragma unroll
    for (int xt = 0; xt < 2; ++xt) {    // x tile (16 pixels each)
      const int xpix = xt * 16 + lr;
      UF a0, a1;
#pragma unroll
      for (int j = 0; j < 16; ++j) {
        int k = (j < 8) ? (kbA + j) : (16 + kbA + (j - 8));
        _Float16 v0 = (_Float16)0.f, v1 = (_Float16)0.f;
        if (k < 27) {
          int dy = k / 9, rr = k % 9, dx = rr / 3, c = rr - dx * 3;
          v0 = lX[((y0 + dy) * 34 + (xpix + dx)) * 3 + c];
          v1 = lX[((y0 + 1 + dy) * 34 + (xpix + dx)) * 3 + c];
        }
        a0.h[j] = v0;
        a1.h[j] = v1;
      }
#pragma unroll
      for (int nt = 0; nt < 2; ++nt) {
        v8f acc0 = vzero8(), acc1 = vzero8();
        acc0 = __builtin_amdgcn_wmma_f32_16x16x32_f16(false, a0.v, false, bf[nt].v,
                                                      (short)0, acc0, false, false);
        acc1 = __builtin_amdgcn_wmma_f32_16x16x32_f16(false, a1.v, false, bf[nt].v,
                                                      (short)0, acc1, false, false);
        const int ch = nt * 16 + lr;
        const float bia = bias[ch];
        const float s = g[ch] * rsqrtf(bv[ch] + BN_EPS);
        const float tt = bb[ch] - bm[ch] * s;
#pragma unroll
        for (int mm = 0; mm < 4; ++mm) {
          float q00 = fmaxf(acc0[2 * mm] + bia, 0.f) * s + tt;
          float q01 = fmaxf(acc0[2 * mm + 1] + bia, 0.f) * s + tt;
          float q10 = fmaxf(acc1[2 * mm] + bia, 0.f) * s + tt;
          float q11 = fmaxf(acc1[2 * mm + 1] + bia, 0.f) * s + tt;
          float pm = fmaxf(fmaxf(q00, q01), fmaxf(q10, q11));
          int py = wv * 2 + p;
          int px = xt * 8 + mm + half * 4;
          h1[(((size_t)img * 16 + py) * 16 + px) * 32 + ch] = (_Float16)pm;
        }
      }
    }
  }
}

// ---------------------------------------------------------------------------
// Kernel 2: conv2 (3x3, 32->64) + bias + relu + BN + pool -> h2 [2048,4096] f16
// One block per image. Input patch staged via async global->LDS B128.
// ---------------------------------------------------------------------------
__global__ __launch_bounds__(256) void conv2_kernel(
    const _Float16* __restrict__ h1, const float* __restrict__ w2,
    const float* __restrict__ bias, const float* __restrict__ g,
    const float* __restrict__ bb, const float* __restrict__ bm,
    const float* __restrict__ bv, _Float16* __restrict__ h2) {
  __shared__ _Float16 lH[18 * 18 * 32];   // padded input patch
  __shared__ _Float16 lWt[64 * 288];      // weights transposed [o][k]

  const int tid = threadIdx.x;
  const int img = blockIdx.x;

  for (int i = tid; i < 18 * 18 * 32; i += 256) lH[i] = (_Float16)0.f;
  __syncthreads();

  // interior: 256 pixels x 64B, async copy (16B per lane, all aligned)
  {
    const void* src = (const void*)(h1 + (size_t)img * 16 * 16 * 32);
    const u32 lbase = lds_off(lH);
#pragma unroll
    for (int i = 0; i < 4; ++i) {
      int d = tid + i * 256;          // 1024 chunks of 16B
      int p = d >> 2, chunk = d & 3;
      int yy = p >> 4, xx = p & 15;
      u32 ldst = lbase + (u32)((((yy + 1) * 18 + (xx + 1)) * 32) * 2 + chunk * 16);
      u32 gsrc = (u32)(p * 64 + chunk * 16);
      async_copy_b128(ldst, gsrc, src);
    }
  }
  // cW2 is HWIO [3,3,32,64] -> flat = k*64 + o with k = (dy*3+dx)*32 + c
  for (int i = tid; i < 288 * 64; i += 256) {
    int k = i >> 6, o = i & 63;
    lWt[o * 288 + k] = (_Float16)w2[i];
  }
  wait_async0();
  __syncthreads();

  const int wv = tid >> 5, lane = tid & 31;
  const int lr = lane & 15, half = lane >> 4;
  const int kbA = half ? 8 : 0;
  const int kbB = half ? 16 : 0;
  const int y0 = 2 * wv;

  v8f acc[2][4];
#pragma unroll
  for (int yy = 0; yy < 2; ++yy)
#pragma unroll
    for (int nt = 0; nt < 4; ++nt) acc[yy][nt] = vzero8();

  for (int ks = 0; ks < 9; ++ks) {
    const int dy = ks / 3, dx = ks % 3;
    UF a[2];
#pragma unroll
    for (int yy = 0; yy < 2; ++yy) {
      int base = ((y0 + yy + dy) * 18 + (lr + dx)) * 32;
#pragma unroll
      for (int t = 0; t < 8; ++t) {
        int c0 = (t < 4) ? (kbA + 2 * t) : (16 + kbA + 2 * (t - 4));
        a[yy].u[t] = *(const u32*)&lH[base + c0];
      }
    }
#pragma unroll
    for (int nt = 0; nt < 4; ++nt) {
      UF b;
      int o = nt * 16 + lr;
      int kst = ks * 32 + kbB;
#pragma unroll
      for (int t = 0; t < 8; ++t)
        b.u[t] = *(const u32*)&lWt[o * 288 + ks * 32 + ((t < 4) ? (kbA + 2 * t) : (16 + kbA + 2 * (t - 4)))];
#pragma unroll
      for (int yy = 0; yy < 2; ++yy)
        acc[yy][nt] = __builtin_amdgcn_wmma_f32_16x16x32_f16(
            false, a[yy].v, false, b.v, (short)0, acc[yy][nt], false, false);
    }
  }

#pragma unroll
  for (int nt = 0; nt < 4; ++nt) {
    const int ch = nt * 16 + lr;
    const float bia = bias[ch];
    const float s = g[ch] * rsqrtf(bv[ch] + BN_EPS);
    const float tt = bb[ch] - bm[ch] * s;
#pragma unroll
    for (int mm = 0; mm < 4; ++mm) {
      float q00 = fmaxf(acc[0][nt][2 * mm] + bia, 0.f) * s + tt;
      float q01 = fmaxf(acc[0][nt][2 * mm + 1] + bia, 0.f) * s + tt;
      float q10 = fmaxf(acc[1][nt][2 * mm] + bia, 0.f) * s + tt;
      float q11 = fmaxf(acc[1][nt][2 * mm + 1] + bia, 0.f) * s + tt;
      float pm = fmaxf(fmaxf(q00, q01), fmaxf(q10, q11));
      int py = wv;
      int px = mm + half * 4;
      h2[(size_t)img * 4096 + (py * 8 + px) * 64 + ch] = (_Float16)pm;
    }
  }
}

// ---------------------------------------------------------------------------
// Generic WMMA GEMM: out = epilogue(A[f16, MxK] @ B[f32, KxNg] + bias)
// Block tile: 128 rows x (NT*16) cols, 8 waves (wave w owns m-tile w).
// A tile staged via async global->LDS B128; B staged with f32->f16 convert.
// blockIdx.y indexes experts (strided operands).
// ---------------------------------------------------------------------------
#define LDA 40   // halves; 80B row stride: 16B-aligned, conflict-free gather

template <int NT, bool RELU, bool BN, bool OUTF16>
__global__ __launch_bounds__(256) void gemm_wmma_kernel(
    const _Float16* __restrict__ A, long aStrideE,
    const float* __restrict__ Bm, long bStrideE,
    const float* __restrict__ bias, long biasStrideE,
    const float* __restrict__ bng, const float* __restrict__ bnb,
    const float* __restrict__ bnm, const float* __restrict__ bnv, long bnStrideE,
    void* __restrict__ outP, long outStrideE, int M, int K, int Ng) {
  constexpr int N16 = NT * 16;
  __shared__ _Float16 lA[128 * LDA];
  __shared__ _Float16 lBt[N16 * 34];

  const int tid = threadIdx.x;
  const int e = blockIdx.y;
  const _Float16* Ae = A + (size_t)e * aStrideE;
  const float* Be = Bm + (size_t)e * bStrideE;
  const float* bi = bias + (size_t)e * biasStrideE;

  const int wv = tid >> 5, lane = tid & 31;
  const int lr = lane & 15, half = lane >> 4;
  const int kbA = half ? 8 : 0;
  const int kbB = half ? 16 : 0;
  const int mbase = blockIdx.x * 128;
  const u32 lAbase = lds_off(lA);

  v8f acc[NT];
#pragma unroll
  for (int nt = 0; nt < NT; ++nt) acc[nt] = vzero8();

  const int ksteps = K >> 5;
  for (int ks = 0; ks < ksteps; ++ks) {
    __syncthreads();
    // stage A tile (128 x 32 f16) via async DMA: 512 x 16B chunks
#pragma unroll
    for (int i = 0; i < 2; ++i) {
      int d = tid + i * 256;
      int mR = d >> 2, chunk = d & 3;
      u32 ldst = lAbase + (u32)(mR * (LDA * 2) + chunk * 16);
      u32 gsrc = (u32)(((size_t)(mbase + mR) * K + ks * 32) * 2 + chunk * 16);
      async_copy_b128(ldst, gsrc, (const void*)Ae);
    }
    // prefetch next B panel while staging this one
    if (ks + 1 < ksteps)
      __builtin_prefetch(&Be[(size_t)((ks + 1) * 32) * Ng], 0, 0);
    // stage B tile transposed [n][k], converting f32->f16, zero-pad n>=Ng
    for (int i = tid; i < 32 * N16; i += 256) {
      int kk = i / N16, n = i % N16;
      float f = (n < Ng) ? Be[(size_t)(ks * 32 + kk) * Ng + n] : 0.f;
      lBt[n * 34 + kk] = (_Float16)f;
    }
    wait_async0();
    __syncthreads();

    UF a;
    const int mR = wv * 16 + lr;
#pragma unroll
    for (int t = 0; t < 8; ++t) {
      int k0 = (t < 4) ? (kbA + 2 * t) : (16 + kbA + 2 * (t - 4));
      a.u[t] = *(const u32*)&lA[mR * LDA + k0];
    }
#pragma unroll
    for (int nt = 0; nt < NT; ++nt) {
      UF b;
      int n = nt * 16 + lr;
#pragma unroll
      for (int t = 0; t < 8; ++t)
        b.u[t] = *(const u32*)&lBt[n * 34 + ((t < 4) ? (kbA + 2 * t) : (16 + kbA + 2 * (t - 4)))];
      acc[nt] = __builtin_amdgcn_wmma_f32_16x16x32_f16(
          false, a.v, false, b.v, (short)0, acc[nt], false, false);
    }
  }

#pragma unroll
  for (int nt = 0; nt < NT; ++nt) {
    const int col = nt * 16 + lr;
    float bv_ = (col < Ng) ? bi[col] : 0.f;
    float s = 1.f, tt = 0.f;
    if constexpr (BN) {
      if (col < Ng) {
        const float* G = bng + (size_t)e * bnStrideE;
        const float* Bb = bnb + (size_t)e * bnStrideE;
        const float* Mn = bnm + (size_t)e * bnStrideE;
        const float* Vv = bnv + (size_t)e * bnStrideE;
        s = G[col] * rsqrtf(Vv[col] + BN_EPS);
        tt = Bb[col] - Mn[col] * s;
      }
    }
#pragma unroll
    for (int r = 0; r < 8; ++r) {
      int row = mbase + wv * 16 + r + half * 8;
      float vv = acc[nt][r] + bv_;
      if constexpr (RELU) vv = fmaxf(vv, 0.f);
      if constexpr (BN) vv = vv * s + tt;
      if (col < Ng) {
        size_t idx = (size_t)e * outStrideE + (size_t)row * Ng + col;
        if constexpr (OUTF16)
          ((_Float16*)outP)[idx] = (_Float16)vv;
        else
          ((float*)outP)[idx] = vv;
      }
    }
  }
}

// ---------------------------------------------------------------------------
// Router: probs = softmax(relu(feats@rW1 + rb1)@rW2 + rb2). One thread/row.
// ---------------------------------------------------------------------------
__global__ __launch_bounds__(256) void router_kernel(
    const _Float16* __restrict__ feats, const float* __restrict__ rW1,
    const float* __restrict__ rb1, const float* __restrict__ rW2,
    const float* __restrict__ rb2, float* __restrict__ probs) {
  __shared__ float sW1[128 * 64];
  __shared__ float sW2[64 * 16];
  const int tid = threadIdx.x;
  for (int i = tid; i < 128 * 64; i += 256) sW1[i] = rW1[i];
  for (int i = tid; i < 64 * 16; i += 256) sW2[i] = rW2[i];
  __syncthreads();

  const int b = blockIdx.x * 256 + tid;
  float h[64];
#pragma unroll
  for (int j = 0; j < 64; ++j) h[j] = rb1[j];
  for (int k = 0; k < 128; ++k) {
    float f = (float)feats[(size_t)b * 128 + k];
#pragma unroll
    for (int j = 0; j < 64; ++j) h[j] = fmaf(f, sW1[k * 64 + j], h[j]);
  }
#pragma unroll
  for (int j = 0; j < 64; ++j) h[j] = fmaxf(h[j], 0.f);

  float lg[16];
#pragma unroll
  for (int c = 0; c < 16; ++c) lg[c] = rb2[c];
#pragma unroll
  for (int j = 0; j < 64; ++j) {
    float f = h[j];
#pragma unroll
    for (int c = 0; c < 16; ++c) lg[c] = fmaf(f, sW2[j * 16 + c], lg[c]);
  }
  float mx = lg[0];
#pragma unroll
  for (int c = 1; c < 16; ++c) mx = fmaxf(mx, lg[c]);
  float sum = 0.f;
#pragma unroll
  for (int c = 0; c < 16; ++c) {
    lg[c] = expf(lg[c] - mx);
    sum += lg[c];
  }
  float inv = 1.f / sum;
#pragma unroll
  for (int c = 0; c < 16; ++c) probs[(size_t)b * 16 + c] = lg[c] * inv;
}

// ---------------------------------------------------------------------------
// Combine: out[b,c] = sum_e probs[b,e] * elog[e,b,c]   (deterministic)
// ---------------------------------------------------------------------------
__global__ __launch_bounds__(256) void combine_kernel(
    const float* __restrict__ probs, const float* __restrict__ elog,
    float* __restrict__ out) {
  int idx = blockIdx.x * 256 + threadIdx.x;
  if (idx >= 2048 * 10) return;
  int b = idx / 10, c = idx % 10;
  float s = 0.f;
#pragma unroll
  for (int e = 0; e < 16; ++e)
    s += probs[(size_t)b * 16 + e] * elog[((size_t)e * 2048 + b) * 10 + c];
  out[idx] = s;
}

// ---------------------------------------------------------------------------
// Launcher
// ---------------------------------------------------------------------------
extern "C" void kernel_launch(void* const* d_in, const int* in_sizes, int n_in,
                              void* d_out, int out_size, void* d_ws,
                              size_t ws_size, hipStream_t stream) {
  const float* x    = (const float*)d_in[0];
  const float* cW1  = (const float*)d_in[1];
  const float* cb1  = (const float*)d_in[2];
  const float* bn1g = (const float*)d_in[3];
  const float* bn1b = (const float*)d_in[4];
  const float* bn1m = (const float*)d_in[5];
  const float* bn1v = (const float*)d_in[6];
  const float* cW2  = (const float*)d_in[7];
  const float* cb2  = (const float*)d_in[8];
  const float* bn2g = (const float*)d_in[9];
  const float* bn2b = (const float*)d_in[10];
  const float* bn2m = (const float*)d_in[11];
  const float* bn2v = (const float*)d_in[12];
  const float* dW   = (const float*)d_in[13];
  const float* db   = (const float*)d_in[14];
  const float* rW1  = (const float*)d_in[15];
  const float* rb1  = (const float*)d_in[16];
  const float* rW2  = (const float*)d_in[17];
  const float* rb2  = (const float*)d_in[18];
  const float* eW1  = (const float*)d_in[19];
  const float* eb1  = (const float*)d_in[20];
  const float* ebng = (const float*)d_in[21];
  const float* ebnb = (const float*)d_in[22];
  const float* ebnm = (const float*)d_in[23];
  const float* ebnv = (const float*)d_in[24];
  const float* eW2  = (const float*)d_in[25];
  const float* eb2  = (const float*)d_in[26];
  const float* eW3  = (const float*)d_in[27];
  const float* eb3  = (const float*)d_in[28];

  char* ws = (char*)d_ws;
  const size_t OFF_H1 = 0;                          // 2048*16*16*32 f16 = 32 MB
  const size_t OFF_H2 = OFF_H1 + (size_t)33554432;  // 2048*4096 f16 = 16 MB
  const size_t OFF_FT = OFF_H2 + (size_t)16777216;  // feats f16 2048*128
  const size_t OFF_PR = OFF_FT + (size_t)524288;    // probs f32 2048*16
  const size_t OFF_T1 = OFF_PR + (size_t)131072;    // t1 f16 16*2048*128
  const size_t OFF_T2 = OFF_T1 + (size_t)8388608;   // t2 f16 16*2048*64
  const size_t OFF_EL = OFF_T2 + (size_t)4194304;   // elog f32 16*2048*10

  _Float16* h1    = (_Float16*)(ws + OFF_H1);
  _Float16* h2    = (_Float16*)(ws + OFF_H2);
  _Float16* feats = (_Float16*)(ws + OFF_FT);
  float*    probs = (float*)(ws + OFF_PR);
  _Float16* t1    = (_Float16*)(ws + OFF_T1);
  _Float16* t2    = (_Float16*)(ws + OFF_T2);
  float*    elog  = (float*)(ws + OFF_EL);

  conv1_kernel<<<2048, 256, 0, stream>>>(x, cW1, cb1, bn1g, bn1b, bn1m, bn1v, h1);
  conv2_kernel<<<2048, 256, 0, stream>>>(h1, cW2, cb2, bn2g, bn2b, bn2m, bn2v, h2);

  // dense: feats = relu(h2 @ dW + db), M=2048 K=4096 N=128
  gemm_wmma_kernel<8, true, false, true><<<dim3(16, 1), 256, 0, stream>>>(
      h2, 0, dW, 0, db, 0, nullptr, nullptr, nullptr, nullptr, 0,
      feats, 0, 2048, 4096, 128);

  router_kernel<<<8, 256, 0, stream>>>(feats, rW1, rb1, rW2, rb2, probs);

  // expert layer 1: relu(feats @ eW1[e] + eb1[e]) then BN -> t1, K=128 N=128
  gemm_wmma_kernel<8, true, true, true><<<dim3(16, 16), 256, 0, stream>>>(
      feats, 0, eW1, (long)128 * 128, eb1, 128, ebng, ebnb, ebnm, ebnv, 128,
      t1, (long)2048 * 128, 2048, 128, 128);

  // expert layer 2: relu(t1 @ eW2[e] + eb2[e]) -> t2, K=128 N=64
  gemm_wmma_kernel<4, true, false, true><<<dim3(16, 16), 256, 0, stream>>>(
      t1, (long)2048 * 128, eW2, (long)128 * 64, eb2, 64,
      nullptr, nullptr, nullptr, nullptr, 0,
      t2, (long)2048 * 64, 2048, 128, 64);

  // expert layer 3: t2 @ eW3[e] + eb3[e] -> elog f32, K=64 N=10 (pad 16)
  gemm_wmma_kernel<1, false, false, false><<<dim3(16, 16), 256, 0, stream>>>(
      t2, (long)2048 * 64, eW3, (long)64 * 10, eb3, 10,
      nullptr, nullptr, nullptr, nullptr, 0,
      elog, (long)2048 * 10, 2048, 64, 10);

  combine_kernel<<<80, 256, 0, stream>>>(probs, elog, (float*)d_out);
}
